// Decoder_86011015069969
// MI455X (gfx1250) — matrix operand order repaired and checked by reading, not compiled, not run on hardware
//
#include <hip/hip_runtime.h>
#include <hip/hip_bf16.h>
#include <math.h>

// Problem constants (match reference).
#define S_NODES   256     // candidate nodes per item
#define E_DIM     256     // embedding dim
#define H_DIM     256     // hidden dim
#define M_MAX     64      // decode steps (num_cell)
#define KP_PITCH  257     // kproj LDS row pitch (floats) -> bank-conflict-free row walk
#define CLIP_C    10.0f

typedef __attribute__((ext_vector_type(16))) __bf16 v16bf;
typedef __attribute__((ext_vector_type(8)))  float  v8f;

// splitmix64-style hash -> uniform (0,1); keyed on (batch, step, node).
__device__ __forceinline__ float uhash(unsigned b, unsigned i, unsigned s) {
    unsigned long long x = ((unsigned long long)b << 40) ^
                           ((unsigned long long)(i + 1) << 20) ^
                           (unsigned long long)(s + 1);
    x ^= x >> 33; x *= 0xff51afd7ed558ccdULL;
    x ^= x >> 33; x *= 0xc4ceb9fe1a85ec53ULL;
    x ^= x >> 33;
    return ((float)((x >> 40) & 0xFFFFFF) + 0.5f) * (1.0f / 16777216.0f);
}

// Saturating fast tanh: no inf/inf NaN for large |x|, 2 transcendentals -> 1.
__device__ __forceinline__ float fast_tanh(float x) {
    const float t = __expf(-2.0f * fabsf(x));
    return copysignf((1.0f - t) / (1.0f + t), x);
}

__global__ __launch_bounds__(256, 1)
void decoder_kernel(const float* __restrict__ cell_embed,   // [B,S,E]
                    const float* __restrict__ original_node,// [B,S,4]
                    const int*   __restrict__ num_cell,     // [1]
                    const float* __restrict__ costs,        // [B,S]
                    const float* __restrict__ init_w,       // [2E]
                    const float* __restrict__ Whc,          // [E,E]
                    const float* __restrict__ bhc,          // [E]
                    const float* __restrict__ Wv,           // [2E,E]
                    const float* __restrict__ bv,           // [E]
                    const float* __restrict__ Wq,           // [E,H]
                    const float* __restrict__ Wk,           // [E,H]
                    const float* __restrict__ vptr,         // [H]
                    float* __restrict__ out,                // [B + B + B*M]
                    int B)
{
    const int b   = blockIdx.x;
    const int tid = threadIdx.x;           // 0..255 ; tid == node index s == dim index e
    const int lane  = tid & 31;
    const int wave  = tid >> 5;
    const int mrow  = lane & 15;           // M/N index inside a 16x16 tile
    const int khalf = lane >> 4;           // 0: low-K half lanes, 1: high-K half lanes

    const float* emb   = cell_embed    + (size_t)b * S_NODES * E_DIM;
    const float* node  = original_node + (size_t)b * S_NODES * 4;
    const float* cost  = costs         + (size_t)b * S_NODES;

    // ---- dynamic LDS carve-up (total ~273 KB < 320 KB WGP LDS) ----
    extern __shared__ float smem[];
    float* kproj = smem;                          // [S][KP_PITCH]  (pitch 257)
    float* qv    = kproj + S_NODES * KP_PITCH;    // [E]   current query
    float* qWq   = qv    + E_DIM;                 // [H]   query @ Wq
    float* hbar  = qWq   + H_DIM;                 // [E]   context embed
    float* inith = hbar  + E_DIM;                 // [E]
    float* hsel  = inith + E_DIM;                 // [E]
    float* lgarr = hsel  + E_DIM;                 // [S]   masked logits (also mean scratch)
    float* redv  = lgarr + S_NODES;               // [256] reduction values
    float* vp    = redv  + 256;                   // [H]   vptr copy
    int*   redi  = (int*)(vp + H_DIM);            // [256] reduction indices
    int*   maskv = redi + 256;                    // [S]

    // ================= Phase A: h_bar = mean(emb) @ Whc + bhc =================
    {
        float msum = 0.0f;
        for (int s = 0; s < S_NODES; ++s) msum += emb[(size_t)s * E_DIM + tid];
        lgarr[tid] = msum * (1.0f / (float)S_NODES);   // reuse lgarr as mean scratch
        __syncthreads();
        float hb = bhc[tid];
        for (int j = 0; j < E_DIM; ++j) hb += lgarr[j] * Whc[(size_t)j * E_DIM + tid];
        hbar[tid] = hb;
        vp[tid]   = vptr[tid];
    }
    __syncthreads();

    // ====== Phase B: kproj = emb @ Wk via v_wmma_f32_16x16x32_bf16 ======
    // 128 tile-pairs: 16 s-tiles x 8 h-tile-pairs; one A tile (16x32) feeds two
    // B tiles (32x16 each) -> 2 WMMA per K-chunk, K fully unrolled (16 WMMA/iter).
    for (int tp = wave; tp < 128; tp += 8) {
        const int s0 = (tp >> 3) << 4;   // s-tile origin
        const int h0 = (tp & 7) << 5;    // h-pair origin (covers h0..h0+31)
        v8f acc0 = {}, acc1 = {};
#pragma unroll
        for (int k0 = 0; k0 < E_DIM; k0 += 32) {
            // ---- A operand: lane needs K = khalf*8+[0..7] and 16+khalf*8+[0..7],
            //      two contiguous 16B-aligned float4-pairs in the emb row.
            const float* arow = emb + (size_t)(s0 + mrow) * E_DIM + k0 + khalf * 8;
            const float4 r0 = *(const float4*)(arow);
            const float4 r1 = *(const float4*)(arow + 4);
            const float4 r2 = *(const float4*)(arow + 16);
            const float4 r3 = *(const float4*)(arow + 20);
            v16bf a;
            a[0]  = (__bf16)r0.x; a[1]  = (__bf16)r0.y; a[2]  = (__bf16)r0.z; a[3]  = (__bf16)r0.w;
            a[4]  = (__bf16)r1.x; a[5]  = (__bf16)r1.y; a[6]  = (__bf16)r1.z; a[7]  = (__bf16)r1.w;
            a[8]  = (__bf16)r2.x; a[9]  = (__bf16)r2.y; a[10] = (__bf16)r2.z; a[11] = (__bf16)r2.w;
            a[12] = (__bf16)r3.x; a[13] = (__bf16)r3.y; a[14] = (__bf16)r3.z; a[15] = (__bf16)r3.w;

            // Prefetch next K-chunk of Wk rows into WGP$ (global_prefetch_b8).
            if (k0 + 32 < E_DIM)
                __builtin_prefetch(Wk + (size_t)(k0 + 32 + khalf * 16) * H_DIM + h0 + mrow, 0, 0);

            // ---- B operands: lanes 0-15 hold K=0..7,16..23, lanes 16-31 K=8..15,24..31 (N=mrow).
            v16bf b0, b1;
#pragma unroll
            for (int e = 0; e < 16; ++e) {
                const size_t roff = (size_t)(k0 + khalf * 8 + e + (e & 8)) * H_DIM + mrow;
                b0[e] = (__bf16)Wk[roff + h0];
                b1[e] = (__bf16)Wk[roff + h0 + 16];
            }
            acc0 = __builtin_amdgcn_wmma_f32_16x16x32_bf16(
                       false, a, false, b0, (short)0, acc0, false, false);
            acc1 = __builtin_amdgcn_wmma_f32_16x16x32_bf16(
                       false, a, false, b1, (short)0, acc1, false, false);
        }
        // D layout: VGPR r holds row M=r (lanes 0-15) / M=r+8 (lanes 16-31), N=lane&15.
#pragma unroll
        for (int r = 0; r < 8; ++r) {
            const int m = r + khalf * 8;
            kproj[(size_t)(s0 + m) * KP_PITCH + (h0 + mrow)]      = acc0[r];
            kproj[(size_t)(s0 + m) * KP_PITCH + (h0 + 16 + mrow)] = acc1[r];
        }
    }

    // ================= Phase C: query at step 0 =================
    {
        float q0 = hbar[tid] + bv[tid];
        for (int j = 0; j < 2 * E_DIM; ++j)
            q0 += init_w[j] * Wv[(size_t)j * E_DIM + tid];
        qv[tid]    = q0;
        maskv[tid] = 0;
        inith[tid] = 0.0f;
    }
    __syncthreads();

    // ================= Phase D: sequential decode loop =================
    const int Msteps = min(num_cell[0], M_MAX);
    float lpsum = 0.0f, rwsum = 0.0f;
    int prev = 0;

    for (int i = 0; i < Msteps; ++i) {
        // (1) qWq[h] = sum_e q[e] * Wq[e][h]    (thread tid == h, coalesced Wq reads)
        {
            float aq = 0.0f;
#pragma unroll 4
            for (int e = 0; e < E_DIM; ++e)
                aq = fmaf(qv[e], Wq[(size_t)e * H_DIM + tid], aq);
            qWq[tid] = aq;
        }
        __syncthreads();

        // (2) u[s] = vptr . tanh(qWq + kproj[s,:]) ; lg = mask ? -1e9 : C*tanh(u)
        //     kproj row walk: pitch 257 -> bank (s+h) mod 64, conflict-free.
        float u = 0.0f;
        {
            const float* krow = kproj + (size_t)tid * KP_PITCH;
#pragma unroll 4
            for (int hh = 0; hh < H_DIM; ++hh)
                u = fmaf(vp[hh], fast_tanh(qWq[hh] + krow[hh]), u);
        }
        const float lg = (maskv[tid] > 0) ? -1e9f : (CLIP_C * fast_tanh(u));
        lgarr[tid] = lg;

        // Gumbel-max categorical sample: argmax(lg + g)
        {
            const float um = uhash((unsigned)b, (unsigned)i, (unsigned)tid);
            redv[tid] = lg + (-logf(-logf(um)));
            redi[tid] = tid;
        }
        __syncthreads();
        for (int off = 128; off > 0; off >>= 1) {
            if (tid < off && redv[tid + off] > redv[tid]) {
                redv[tid] = redv[tid + off];
                redi[tid] = redi[tid + off];
            }
            __syncthreads();
        }
        const int idx = redi[0];
        __syncthreads();

        // (3) log_softmax(lg)[idx] via max + sum-exp reductions
        redv[tid] = lg;
        __syncthreads();
        for (int off = 128; off > 0; off >>= 1) {
            if (tid < off) redv[tid] = fmaxf(redv[tid], redv[tid + off]);
            __syncthreads();
        }
        const float maxlg = redv[0];
        __syncthreads();
        redv[tid] = __expf(lg - maxlg);
        __syncthreads();
        for (int off = 128; off > 0; off >>= 1) {
            if (tid < off) redv[tid] += redv[tid + off];
            __syncthreads();
        }
        const float lse = maxlg + logf(redv[0]);
        __syncthreads();

        // (4) bookkeeping: logp, reward, action (thread 0)
        if (tid == 0) {
            lpsum += lgarr[idx] - lse;
            if (i > 0) {
                const float dx = node[(size_t)idx * 4 + 0] - node[(size_t)prev * 4 + 2];
                const float dy = node[(size_t)idx * 4 + 1] - node[(size_t)prev * 4 + 3];
                rwsum += sqrtf(dx * dx + dy * dy) + cost[prev] + cost[idx];
            }
            out[2 * B + (size_t)b * M_MAX + i] = (float)idx;
        }

        // (5) mask whole 4-node cell of idx; fetch selected embedding row
        {
            const int cell0 = (idx >> 2) << 2;
            if (tid >= cell0 && tid < cell0 + 4) maskv[tid] = 1;
            const float h = emb[(size_t)idx * E_DIM + tid];
            hsel[tid] = h;
            if (i == 0) inith[tid] = h;
        }
        __syncthreads();

        // (6) query = h_bar + concat(init_h, h) @ Wv + bv   (thread tid == out dim)
        {
            float aq = hbar[tid] + bv[tid];
#pragma unroll 4
            for (int j = 0; j < E_DIM; ++j)
                aq = fmaf(inith[j], Wv[(size_t)j * E_DIM + tid], aq);
#pragma unroll 4
            for (int j = 0; j < E_DIM; ++j)
                aq = fmaf(hsel[j], Wv[(size_t)(E_DIM + j) * E_DIM + tid], aq);
            qv[tid] = aq;
        }
        prev = idx;
        __syncthreads();
    }

    if (tid == 0) {
        out[b]     = lpsum;
        out[B + b] = rwsum;
    }
}

extern "C" void kernel_launch(void* const* d_in, const int* in_sizes, int n_in,
                              void* d_out, int out_size, void* d_ws, size_t ws_size,
                              hipStream_t stream) {
    const float* cell_embed    = (const float*)d_in[0];
    const float* original_node = (const float*)d_in[1];
    // d_in[2] = maze (unused by the math)
    const int*   num_cell      = (const int*)d_in[3];
    const float* costs         = (const float*)d_in[4];
    const float* init_w        = (const float*)d_in[5];
    const float* Whc           = (const float*)d_in[6];
    const float* bhc           = (const float*)d_in[7];
    const float* Wv            = (const float*)d_in[8];
    const float* bv            = (const float*)d_in[9];
    const float* Wq            = (const float*)d_in[10];
    const float* Wk            = (const float*)d_in[11];
    const float* vptr          = (const float*)d_in[12];
    float* out = (float*)d_out;

    const int B = in_sizes[0] / (S_NODES * E_DIM);

    // LDS: kproj (256*257) + 8 float arrays (256) + redi/mask ints (512) floats.
    const size_t shmem = (size_t)(S_NODES * KP_PITCH + 8 * 256 + 512) * sizeof(float);

    hipLaunchKernelGGL(decoder_kernel, dim3(B), dim3(256), shmem, stream,
                       cell_embed, original_node, num_cell, costs, init_w,
                       Whc, bhc, Wv, bv, Wq, Wk, vptr, out, B);
}
